// LstmLowMem_20813411517095
// MI455X (gfx1250) — matrix operand order repaired and checked by reading, not compiled, not run on hardware
//
#include <hip/hip_runtime.h>
#include <stdint.h>

// ---------------- Problem constants (from reference) ----------------
#define TSTEPS 512
#define BATCH  64
#define NIN    1024
#define HDIM   1024
#define NG     4096   // 4*H

typedef __attribute__((ext_vector_type(16))) __bf16 v16bf;
typedef __attribute__((ext_vector_type(8)))  float  v8f;

struct BF16x16 { uint4 lo, hi; };   // 32 bytes, reinterpreted as v16bf

// fp32 -> bf16 round-to-nearest-even (raw ushort)
__device__ __forceinline__ unsigned short f2bf(float f) {
    unsigned int u = __builtin_bit_cast(unsigned int, f);
    unsigned int lsb = (u >> 16) & 1u;
    u += 0x7fffu + lsb;
    return (unsigned short)(u >> 16);
}

// ---- CDNA5 async global->LDS copy (ASYNCcnt path) ----
// Builtin signature (from hipcc diagnostic): param0 = v4i addrspace(1)* (global src),
// param1 = LDS dest pointer, params 2/3 = imm offset / imm cpol.
// Generic shared-memory pointer: low 32 bits are the LDS byte address
typedef __attribute__((vector_size(16))) int async_v4i;
typedef __attribute__((address_space(1))) async_v4i* g_v4i_ptr;
typedef __attribute__((address_space(3))) async_v4i* l_v4i_ptr;

__device__ __forceinline__ void async_b128(const void* g, void* l) {
#if __has_builtin(__builtin_amdgcn_global_load_async_to_lds_b128)
    __builtin_amdgcn_global_load_async_to_lds_b128(
        (g_v4i_ptr)(unsigned long long)(uintptr_t)g,
        (l_v4i_ptr)(unsigned)(uintptr_t)l,
        0, 0);
#else
    unsigned lds_off = (unsigned)(uintptr_t)l;
    unsigned long long ga = (unsigned long long)(uintptr_t)g;
    asm volatile("global_load_async_to_lds_b128 %0, %1, off"
                 :: "v"(lds_off), "v"(ga) : "memory");
#endif
}

__device__ __forceinline__ void wait_async0() {
#if __has_builtin(__builtin_amdgcn_s_wait_asynccnt)
    __builtin_amdgcn_s_wait_asynccnt(0);
#else
    asm volatile("s_wait_asynccnt 0" ::: "memory");
#endif
}

// A-matrix 16x32 bf16 fragment per ISA layout:
// lane<16 : K runs [0..7],[16..23]; lane>=16: [8..15],[24..31]  (row M = lane&15)
__device__ __forceinline__ v16bf ld_a16(const unsigned short* __restrict__ rowp, int k0, int lane) {
    const int b0 = (lane & 16) ? 8 : 0;
    BF16x16 t;
    t.lo = *(const uint4*)(rowp + k0 + b0);
    t.hi = *(const uint4*)(rowp + k0 + b0 + 16);
    return __builtin_bit_cast(v16bf, t);
}

// B-matrix 32x16 bf16 fragment from col-major (N x K) source:
// lanes<16 hold K=[0..7],[16..23]; lanes>=16 hold K=[8..15],[24..31] (matches A).
__device__ __forceinline__ v16bf ld_b16(const unsigned short* __restrict__ colp, int k0, int lane) {
    const int kb = (lane & 16) ? 8 : 0;
    BF16x16 t;
    t.lo = *(const uint4*)(colp + k0 + kb);
    t.hi = *(const uint4*)(colp + k0 + kb + 16);
    return __builtin_bit_cast(v16bf, t);
}

// ---------------- Conversion / init kernels ----------------
__global__ void k_cvt_x(const float* __restrict__ x, unsigned short* __restrict__ o, int n) {
    for (int i = blockIdx.x * blockDim.x + threadIdx.x; i < n; i += gridDim.x * blockDim.x)
        o[i] = f2bf(x[i]);
}

__global__ void k_cvt_w(const float* __restrict__ W,
                        unsigned short* __restrict__ WxT,
                        unsigned short* __restrict__ WhT) {
    const int n = NG * NIN;
    for (int i = blockIdx.x * blockDim.x + threadIdx.x; i < n; i += gridDim.x * blockDim.x) {
        int ncol = i >> 10;
        int k    = i & 1023;
        WxT[i] = f2bf(W[(size_t)k * NG + ncol]);
        WhT[i] = f2bf(W[(size_t)(NIN + k) * NG + ncol]);
    }
}

__global__ void k_init(const float* __restrict__ y0, const float* __restrict__ c0,
                       unsigned short* __restrict__ ybf, float* __restrict__ cst,
                       unsigned int* __restrict__ bar) {
    int i = blockIdx.x * blockDim.x + threadIdx.x;
    if (i < BATCH * HDIM) {
        ybf[i] = f2bf(y0[i]);
        cst[i] = c0[i];
    }
    if (i == 0) *bar = 0u;
}

// ---------------- Phase 1: Zx = Xbf @ Wx + b ----------------
// Block 256 thr (8 waves), block tile 128x128, KC=64 K-chunk.
// Double-buffered LDS tiles filled by async global->LDS b128 copies; rows padded
// to KCP=72 halfwords so the 16-row fragment reads hit 16 distinct LDS banks.
#define KC  64
#define KCP 72
#define TILE_BYTES (128 * KCP * 2)         // 18432 B per tile
#define ZX_LDS_BYTES (4 * TILE_BYTES)      // 2 stages x (A+B) = 73728 B

__global__ void __launch_bounds__(256)
k_zx_gemm(const unsigned short* __restrict__ Xbf,
          const unsigned short* __restrict__ WxT,
          const float* __restrict__ bias,
          float* __restrict__ Zx) {
    extern __shared__ char smem[];
    // Stage bases computed arithmetically (no pointer aggregates of shared mem):
    //   A stage st: smem + st*TILE_BYTES ; B stage st: smem + (2+st)*TILE_BYTES

    const int tid = threadIdx.x, wave = tid >> 5, lane = tid & 31;
    const int wm = wave & 3, wn = wave >> 2;       // wave tile: 32(M) x 64(N)
    const int m0 = blockIdx.x * 128;
    const int n0 = blockIdx.y * 128;
    const int nl = lane & 15;

    // Each thread issues 4 A + 4 B async b128 transfers per stage (16 KB each tile).
    auto stage_copy = [&](int st, int kc) {
        const unsigned short* gA = Xbf + (size_t)m0 * NIN + kc * KC;
        const unsigned short* gB = WxT + (size_t)n0 * NIN + kc * KC;
        unsigned short* dA = (unsigned short*)(smem + st * TILE_BYTES);
        unsigned short* dB = (unsigned short*)(smem + (2 + st) * TILE_BYTES);
        for (int q = tid; q < 1024; q += 256) {
            const int row = q >> 3, ch = (q & 7) * 8;
            async_b128(gA + (size_t)row * NIN + ch, dA + row * KCP + ch);
            async_b128(gB + (size_t)row * NIN + ch, dB + row * KCP + ch);
        }
    };

    v8f acc[2][4];
#pragma unroll
    for (int u = 0; u < 4; ++u) {
        const float bv = bias[n0 + wn * 64 + u * 16 + nl];
#pragma unroll
        for (int s = 0; s < 2; ++s)
#pragma unroll
            for (int r = 0; r < 8; ++r) acc[s][u][r] = bv;
    }

    stage_copy(0, 0);
    wait_async0();
    __syncthreads();

    const int nk = NIN / KC;   // 16
    for (int kc = 0; kc < nk; ++kc) {
        const int cur = kc & 1;
        if (kc + 1 < nk) stage_copy(cur ^ 1, kc + 1);

        const unsigned short* Ab = (const unsigned short*)(smem + cur * TILE_BYTES);
        const unsigned short* Bb = (const unsigned short*)(smem + (2 + cur) * TILE_BYTES);
#pragma unroll
        for (int k0 = 0; k0 < KC; k0 += 32) {
            v16bf am[2], bm[4];
#pragma unroll
            for (int s = 0; s < 2; ++s)
                am[s] = ld_a16(Ab + (wm * 32 + s * 16 + nl) * KCP, k0, lane);
#pragma unroll
            for (int u = 0; u < 4; ++u)
                bm[u] = ld_b16(Bb + (wn * 64 + u * 16 + nl) * KCP, k0, lane);
#pragma unroll
            for (int s = 0; s < 2; ++s)
#pragma unroll
                for (int u = 0; u < 4; ++u)
                    acc[s][u] = __builtin_amdgcn_wmma_f32_16x16x32_bf16(
                        false, am[s], false, bm[u], (short)0, acc[s][u], false, false);
        }

        if (kc + 1 < nk) wait_async0();
        __syncthreads();
    }

    const int rbase = (lane & 16) ? 8 : 0;
#pragma unroll
    for (int s = 0; s < 2; ++s)
#pragma unroll
        for (int u = 0; u < 4; ++u) {
            const int col = n0 + wn * 64 + u * 16 + nl;
#pragma unroll
            for (int r = 0; r < 8; ++r)
                Zx[(size_t)(m0 + wm * 32 + s * 16 + rbase + r) * NG + col] = acc[s][u][r];
        }
}

// ---------------- Phase 2: persistent recurrent kernel ----------------
// 64 blocks; block j owns h-columns [j*16, j*16+16). Wh slice (128 KB bf16)
// async-copied into LDS once and pinned for the whole 512-step scan.
__global__ void __launch_bounds__(256)
k_lstm_recur(const float* __restrict__ Zx,
             const unsigned short* __restrict__ WhT,
             const float* __restrict__ iMask,
             float* __restrict__ Y, float* __restrict__ Cout, float* __restrict__ cT,
             unsigned short* __restrict__ ybf,      // 2 buffers of B*H bf16
             float* __restrict__ cstate,
             unsigned int* __restrict__ bar) {
    extern __shared__ char smem[];
    unsigned short* whLds = (unsigned short*)smem;          // 4*16*1024 bf16 = 128 KB
    float*          zLds  = (float*)(smem + 131072);        // 4*64*16 f32  =  16 KB

    const int j = blockIdx.x;
    const int tid = threadIdx.x, wave = tid >> 5, lane = tid & 31;
    const int g = wave >> 1, nl = lane & 15;
    const int rbase = (lane & 16) ? 8 : 0;

    // One-time Wh slice preload: 8192 async b128 transfers (128 KB).
    for (int q = tid; q < 8192; q += 256) {
        const int gg = q >> 11;          // gate
        const int r  = (q & 2047) * 8;   // halfword offset inside 32 KB gate slab
        async_b128(WhT + ((size_t)(gg * HDIM + j * 16)) * HDIM + r,
                   whLds + gg * 16 * 1024 + r);
    }
    wait_async0();
    __syncthreads();

    const unsigned short* bcol = whLds + (size_t)(g * 16 + nl) * 1024;
    const int zcol = g * HDIM + j * 16 + nl;

    for (int t = 0; t < TSTEPS; ++t) {
        const unsigned short* yrd = ybf + (size_t)(t & 1) * (BATCH * HDIM);
        unsigned short*       ywr = ybf + (size_t)((t + 1) & 1) * (BATCH * HDIM);

        v8f acc[2];
#pragma unroll
        for (int s = 0; s < 2; ++s) {
            const int mtile = (wave & 1) * 2 + s;
#pragma unroll
            for (int r = 0; r < 8; ++r)
                acc[s][r] = Zx[((size_t)t * BATCH + mtile * 16 + rbase + r) * NG + zcol];
            if (t + 1 < TSTEPS)  // warm L2 for next step's Zx tile
                __builtin_prefetch(&Zx[((size_t)(t + 1) * BATCH + mtile * 16 + rbase) * NG + zcol], 0, 1);
        }

#pragma unroll 4
        for (int k0 = 0; k0 < HDIM; k0 += 32) {
            v16bf bm = ld_b16(bcol, k0, lane);   // LDS-resident Wh
#pragma unroll
            for (int s = 0; s < 2; ++s) {
                const int mtile = (wave & 1) * 2 + s;
                const unsigned short* arow = yrd + (size_t)(mtile * 16 + nl) * HDIM;
                v16bf am = ld_a16(arow, k0, lane);
                acc[s] = __builtin_amdgcn_wmma_f32_16x16x32_bf16(
                    false, am, false, bm, (short)0, acc[s], false, false);
            }
        }

#pragma unroll
        for (int s = 0; s < 2; ++s) {
            const int mtile = (wave & 1) * 2 + s;
#pragma unroll
            for (int r = 0; r < 8; ++r)
                zLds[(g * 64 + mtile * 16 + rbase + r) * 16 + nl] = acc[s][r];
        }
        __syncthreads();

        for (int e = tid; e < BATCH * 16; e += 256) {
            const int row = e >> 4, n = e & 15;
            const float zc = zLds[(0 * 64 + row) * 16 + n];
            const float zi = zLds[(1 * 64 + row) * 16 + n];
            const float zf = zLds[(2 * 64 + row) * 16 + n];
            const float zo = zLds[(3 * 64 + row) * 16 + n];
            const float ci = tanhf(zc);
            const float ig = 1.f / (1.f + __expf(-zi));
            const float fg = 1.f / (1.f + __expf(-zf));
            const float og = 1.f / (1.f + __expf(-zo));
            const int h = j * 16 + n;
            const size_t sidx = (size_t)row * HDIM + h;
            const float cprev = cstate[sidx];
            float c = ci * ig + cprev * fg;
            float y = tanhf(c) * og;
            const float m = iMask[t * BATCH + row];
            c = m * c + (1.f - m) * cprev;
            y = m * y;
            const size_t oidx = ((size_t)t * BATCH + row) * HDIM + h;
            Y[oidx] = y;
            Cout[oidx] = c;
            cstate[sidx] = c;
            ywr[sidx] = f2bf(y);
            if (t == TSTEPS - 1) cT[sidx] = c;
        }
        __syncthreads();

        if (tid == 0) {
            __hip_atomic_fetch_add(bar, 1u, __ATOMIC_RELEASE, __HIP_MEMORY_SCOPE_AGENT);
            const unsigned target = (unsigned)(t + 1) * gridDim.x;
            while (__hip_atomic_load(bar, __ATOMIC_ACQUIRE, __HIP_MEMORY_SCOPE_AGENT) < target)
                __builtin_amdgcn_s_sleep(1);
        }
        __syncthreads();
    }
}

// ---------------- Host launcher ----------------
extern "C" void kernel_launch(void* const* d_in, const int* in_sizes, int n_in,
                              void* d_out, int out_size, void* d_ws, size_t ws_size,
                              hipStream_t stream) {
    const float* X  = (const float*)d_in[0];
    const float* W  = (const float*)d_in[1];
    const float* b  = (const float*)d_in[2];
    const float* y0 = (const float*)d_in[3];
    const float* c0 = (const float*)d_in[4];
    const float* iM = (const float*)d_in[5];

    float* Y    = (float*)d_out;
    float* Cout = Y + (size_t)TSTEPS * BATCH * HDIM;
    float* cT   = Cout + (size_t)TSTEPS * BATCH * HDIM;

    char* ws = (char*)d_ws;
    unsigned short* Xbf = (unsigned short*)(ws);                  // 64 MB
    unsigned short* WxT = (unsigned short*)(ws + 67108864ull);    //  8 MB
    unsigned short* WhT = (unsigned short*)(ws + 75497472ull);    //  8 MB
    float*          Zx  = (float*)(ws + 83886080ull);             // 512 MB
    unsigned short* ybf = (unsigned short*)(ws + 620756992ull);   // 256 KB (2 bufs)
    float*          cst = (float*)(ws + 621019136ull);            // 256 KB
    unsigned int*   bar = (unsigned int*)(ws + 621281280ull);     // 4 B

    k_cvt_x<<<4096, 256, 0, stream>>>(X, Xbf, TSTEPS * BATCH * NIN);
    k_cvt_w<<<4096, 256, 0, stream>>>(W, WxT, WhT);
    k_init<<<(BATCH * HDIM + 255) / 256, 256, 0, stream>>>(y0, c0, ybf, cst, bar);

    (void)hipFuncSetAttribute(reinterpret_cast<const void*>(k_zx_gemm),
                              hipFuncAttributeMaxDynamicSharedMemorySize, ZX_LDS_BYTES);
    k_zx_gemm<<<dim3((TSTEPS * BATCH) / 128, NG / 128), 256, ZX_LDS_BYTES, stream>>>(Xbf, WxT, b, Zx);

    (void)hipFuncSetAttribute(reinterpret_cast<const void*>(k_lstm_recur),
                              hipFuncAttributeMaxDynamicSharedMemorySize, 147456);
    k_lstm_recur<<<HDIM / 16, 256, 147456, stream>>>(Zx, WhT, iM, Y, Cout, cT, ybf, cst, bar);
}
